// UMT_MVSNet_V3_46797963657721
// MI455X (gfx1250) — matrix operand order repaired and checked by reading, not compiled, not run on hardware
//
#include <hip/hip_runtime.h>

// Problem constants
#define BB   2
#define VV   4
#define HH   256
#define WW   256
#define DD   64
#define HWp  65536
#define HIDN 768
#define NHN  8
#define HDN  96

typedef __attribute__((ext_vector_type(16))) _Float16 f16x16;
typedef __attribute__((ext_vector_type(8)))  _Float16 f16x8;
typedef __attribute__((ext_vector_type(8)))  float    f32x8;
typedef __attribute__((ext_vector_type(4)))  int      i32x4;
typedef __attribute__((address_space(1))) i32x4 as1_i32x4;
typedef __attribute__((address_space(3))) i32x4 as3_i32x4;

#define GF_BIAS 1
#define GF_RELU 2
#define GF_BETA 4

// Async global->LDS path (guarded: falls back to plain LDS stores if absent)
#if defined(__gfx1250__) && __has_builtin(__builtin_amdgcn_global_load_async_to_lds_b128) && __has_builtin(__builtin_amdgcn_s_wait_asynccnt)
#define HAS_ASYNC_LDS 1
#else
#define HAS_ASYNC_LDS 0
#endif

// ---------------------------------------------------------------------------
// 1) Per (b,v) homography: proj = src_proj @ inv(ref_proj); store rot(3x3)+t(3)
// ---------------------------------------------------------------------------
__global__ void k_proj(const float* __restrict__ pm, float* __restrict__ prt) {
  int t = threadIdx.x;
  if (t >= BB * VV) return;
  int b = t / VV;
  const float* ref = pm + (size_t)(b * VV + 0) * 16;
  const float* src = pm + (size_t)t * 16;
  float a[4][4], inv[4][4];
  for (int i = 0; i < 4; ++i)
    for (int j = 0; j < 4; ++j) { a[i][j] = ref[i*4+j]; inv[i][j] = (i==j)?1.f:0.f; }
  for (int c = 0; c < 4; ++c) {
    int piv = c; float best = fabsf(a[c][c]);
    for (int r = c+1; r < 4; ++r) { float v = fabsf(a[r][c]); if (v > best) { best = v; piv = r; } }
    if (piv != c)
      for (int j = 0; j < 4; ++j) {
        float tA = a[c][j]; a[c][j] = a[piv][j]; a[piv][j] = tA;
        float tI = inv[c][j]; inv[c][j] = inv[piv][j]; inv[piv][j] = tI;
      }
    float ip = 1.f / a[c][c];
    for (int j = 0; j < 4; ++j) { a[c][j] *= ip; inv[c][j] *= ip; }
    for (int r = 0; r < 4; ++r) if (r != c) {
      float f = a[r][c];
      for (int j = 0; j < 4; ++j) { a[r][j] -= f * a[c][j]; inv[r][j] -= f * inv[c][j]; }
    }
  }
  for (int i = 0; i < 3; ++i)
    for (int j = 0; j < 4; ++j) {
      float s = 0.f;
      for (int kk = 0; kk < 4; ++kk) s += src[i*4+kk] * inv[kk][j];
      prt[t*12 + i*4 + j] = s;
    }
}

// ---------------------------------------------------------------------------
// 2) Warp src view to ref grid per depth, squared diff -> xbuf[(d*B+b),c,pix]
// ---------------------------------------------------------------------------
__global__ __launch_bounds__(256)
void k_warpdiff(const float* __restrict__ imgs, const float* __restrict__ prt,
                const float* __restrict__ dvv, float* __restrict__ xb, int v) {
  int pix = blockIdx.x * 256 + threadIdx.x;
  int n = blockIdx.y;            // n = d*B + b
  int d = n / BB, b = n % BB;
  int yy = pix >> 8, xx = pix & 255;
  const float* rt = prt + (size_t)(b * VV + v) * 12;
  float dep = dvv[b * DD + d];
  float X = (float)xx, Y = (float)yy;
  float p0 = (rt[0]*X + rt[1]*Y + rt[2])  * dep + rt[3];
  float p1 = (rt[4]*X + rt[5]*Y + rt[6])  * dep + rt[7];
  float p2 = (rt[8]*X + rt[9]*Y + rt[10]) * dep + rt[11];
  float gx = p0 / p2, gy = p1 / p2;
  float fx = floorf(gx), fy = floorf(gy);
  float wx = gx - fx, wy = gy - fy;
  int x0 = (int)fx, y0 = (int)fy, x1 = x0 + 1, y1 = y0 + 1;
  float m00 = (x0>=0 && x0<WW && y0>=0 && y0<HH) ? 1.f : 0.f;
  float m10 = (x1>=0 && x1<WW && y0>=0 && y0<HH) ? 1.f : 0.f;
  float m01 = (x0>=0 && x0<WW && y1>=0 && y1<HH) ? 1.f : 0.f;
  float m11 = (x1>=0 && x1<WW && y1>=0 && y1<HH) ? 1.f : 0.f;
  int cx0 = min(max(x0,0),WW-1), cx1 = min(max(x1,0),WW-1);
  int cy0 = min(max(y0,0),HH-1), cy1 = min(max(y1,0),HH-1);
  int i00 = cy0*WW+cx0, i10 = cy0*WW+cx1, i01 = cy1*WW+cx0, i11 = cy1*WW+cx1;
  float w00 = (1.f-wx)*(1.f-wy)*m00, w10 = wx*(1.f-wy)*m10;
  float w01 = (1.f-wx)*wy*m01,       w11 = wx*wy*m11;
  for (int c = 0; c < 3; ++c) {
    const float* sp = imgs + ((size_t)((b*VV+v)*3 + c)) * HWp;
    float wval = sp[i00]*w00 + sp[i10]*w10 + sp[i01]*w01 + sp[i11]*w11;
    float rval = imgs[((size_t)((b*VV)*3 + c)) * HWp + pix];
    float df = wval - rval;
    xb[((size_t)(n*3 + c)) * HWp + pix] = df * df;
  }
}

// ---------------------------------------------------------------------------
// 3) Fused gate: g = sigmoid(conv2(relu(conv1(x)))); acc (+)= (g+1)*x
// ---------------------------------------------------------------------------
__global__ __launch_bounds__(256)
void k_gate(const float* __restrict__ xb,
            const float* __restrict__ w1, const float* __restrict__ b1,
            const float* __restrict__ w2, const float* __restrict__ b2,
            float* __restrict__ accv, int first) {
  __shared__ float w1s[216], w2s[216], b1s[8], b2s[3];
  __shared__ float xt[3][20][20];
  __shared__ float ht[8][18][18];
  int tid = threadIdx.x;
  if (tid < 216) { w1s[tid] = w1[tid]; w2s[tid] = w2[tid]; }
  if (tid < 8)  b1s[tid] = b1[tid];
  if (tid < 3)  b2s[tid] = b2[tid];
  int n   = blockIdx.y;
  int tx0 = (blockIdx.x & 15) * 16;
  int ty0 = (blockIdx.x >> 4) * 16;
  for (int i = tid; i < 1200; i += 256) {
    int c = i / 400, rr = (i % 400) / 20, cc = i % 20;
    int gy = ty0 + rr - 2, gx = tx0 + cc - 2;
    float vI = 0.f;
    if (gy >= 0 && gy < HH && gx >= 0 && gx < WW)
      vI = xb[((size_t)(n*3 + c)) * HWp + gy * WW + gx];
    xt[c][rr][cc] = vI;
  }
  __syncthreads();
  for (int i = tid; i < 8*18*18; i += 256) {
    int co = i / 324, p = i % 324, rr = p / 18, cc = p % 18;
    int oy = ty0 + rr - 1, ox = tx0 + cc - 1;
    float hv = 0.f;
    if (oy >= 0 && oy < HH && ox >= 0 && ox < WW) {
      float s = b1s[co];
      for (int ci = 0; ci < 3; ++ci)
        for (int ky = 0; ky < 3; ++ky)
          for (int kx = 0; kx < 3; ++kx)
            s += w1s[((co*3+ci)*3+ky)*3+kx] * xt[ci][rr+ky][cc+kx];
      hv = fmaxf(s, 0.f);
    }
    ht[co][rr][cc] = hv;
  }
  __syncthreads();
  int rr = tid >> 4, cc = tid & 15;
  int oy = ty0 + rr, ox = tx0 + cc;
  for (int co = 0; co < 3; ++co) {
    float s = b2s[co];
    for (int ci = 0; ci < 8; ++ci)
      for (int ky = 0; ky < 3; ++ky)
        for (int kx = 0; kx < 3; ++kx)
          s += w2s[((co*8+ci)*3+ky)*3+kx] * ht[ci][rr+ky][cc+kx];
    float g = 1.f / (1.f + expf(-s));
    float cval = (g + 1.f) * xt[co][rr+2][cc+2];
    size_t o = ((size_t)(n*3 + co)) * HWp + (size_t)oy * WW + ox;
    if (first) accv[o] = cval; else accv[o] += cval;
  }
}

// ---------------------------------------------------------------------------
// 4) Patchify cost volume /3 -> f16 tokens [32768, 768]
// ---------------------------------------------------------------------------
__global__ __launch_bounds__(256)
void k_patchify(const float* __restrict__ accv, _Float16* __restrict__ tokh) {
  size_t i = (size_t)blockIdx.x * 256 + threadIdx.x;     // < 25165824
  int f = (int)(i % 768);
  size_t r = i / 768;
  int t = (int)(r % 256);
  int n = (int)(r / 256);
  int c = f >> 8, p = f & 255, py = p >> 4, px = p & 15;
  int hp = t >> 4, wp = t & 15;
  float v = accv[((size_t)(n*3 + c)) * HWp + (size_t)(hp*16 + py) * WW + (wp*16 + px)] * (1.f/3.f);
  tokh[i] = (_Float16)v;
}

// ---------------------------------------------------------------------------
// 5a) f32 -> f16 convert (row-major, for activations)
// ---------------------------------------------------------------------------
__global__ __launch_bounds__(256)
void k_f2h(const float* __restrict__ s, _Float16* __restrict__ d, long n) {
  long i = (long)blockIdx.x * 256 + threadIdx.x;
  if (i < n) d[i] = (_Float16)s[i];
}

// 5b) f32 -> f16 pack B into WMMA-fragment-native panels:
//     Bp[((k>>4)*N + n)*16 + (k&15)] = B[k*N + n]
//     => each lane's 16 K-values for a column are contiguous (one 32B load).
__global__ __launch_bounds__(256)
void k_packB(const float* __restrict__ src, _Float16* __restrict__ dst,
             int K, int N) {
  long idx = (long)blockIdx.x * 256 + threadIdx.x;
  if (idx >= (long)K * N) return;
  int k = (int)(idx / N), n = (int)(idx % N);
  dst[(((size_t)(k >> 4)) * N + n) * 16 + (k & 15)] = (_Float16)src[idx];
}

// ---------------------------------------------------------------------------
// 6) WMMA GEMM: Cout[M,N] = A[M,K](f16,row-major) * Bp[K,N](f16,packed panels)
//    4 waves/block; each wave owns a 16x32 output tile (2 WMMAs / K-step,
//    A fragment reused). Fragment k order: contiguous k = 16*lh + s in both A and B (wave32).
// ---------------------------------------------------------------------------
__global__ __launch_bounds__(128)
void k_gemm(const _Float16* __restrict__ A, const _Float16* __restrict__ Bp,
            const float* __restrict__ bias, const float* __restrict__ Cin,
            float* __restrict__ Cout, int M, int N, int K, int flags) {
  int wave = threadIdx.x >> 5;
  int lane = threadIdx.x & 31;
  int lh   = lane >> 4;          // lane half
  int l15  = lane & 15;
  int col0 = (blockIdx.x * 4 + wave) * 32;
  int row0 = blockIdx.y * 16;
  if (col0 >= N || row0 >= M) return;   // wave-uniform
  f32x8 acc0 = {}, acc1 = {};
  int m  = row0 + l15;
  int n0 = col0 + l15;
  int n1 = col0 + 16 + l15;
  for (int k0 = 0; k0 < K; k0 += 32) {
    // A 16x32 f16: lanes 0-15 row m hold K {k0..k0+15}; lanes 16-31 +16
    const _Float16* ap = A + (size_t)m * K + (k0 + 16*lh);
    f16x8 alo = *(const f16x8*)(ap);
    f16x8 ahi = *(const f16x8*)(ap + 8);
    f16x16 a;
    #pragma unroll
    for (int i = 0; i < 8; ++i) { a[i] = alo[i]; a[i+8] = ahi[i]; }
    // Packed B: rows k0+16*lh+e (e=0..15) for column n are contiguous.
    size_t kb = (size_t)((k0 >> 4) + lh);
    f16x16 b0 = *(const f16x16*)(Bp + (kb * N + n0) * 16);
    f16x16 b1 = *(const f16x16*)(Bp + (kb * N + n1) * 16);
    __builtin_prefetch(ap + 32, 0, 1);
    __builtin_prefetch(Bp + ((kb + 2) * N + n0) * 16, 0, 1);
    acc0 = __builtin_amdgcn_wmma_f32_16x16x32_f16(false, a, false, b0,
                                                  (short)0, acc0, false, false);
    acc1 = __builtin_amdgcn_wmma_f32_16x16x32_f16(false, a, false, b1,
                                                  (short)0, acc1, false, false);
  }
  #pragma unroll
  for (int g = 0; g < 8; ++g) {
    int mm = row0 + g + 8*lh;
    float v0 = acc0[g], v1 = acc1[g];
    if (flags & GF_BIAS) { v0 += bias[n0]; v1 += bias[n1]; }
    if (flags & GF_RELU) { v0 = fmaxf(v0, 0.f); v1 = fmaxf(v1, 0.f); }
    size_t i0 = (size_t)mm * N + n0;
    size_t i1 = (size_t)mm * N + n1;
    if (flags & GF_BETA) { v0 += Cin[i0]; v1 += Cin[i1]; }
    Cout[i0] = v0;
    Cout[i1] = v1;
  }
}

// ---------------------------------------------------------------------------
// 7) LayerNorm per row (in place, f32)
// ---------------------------------------------------------------------------
__global__ __launch_bounds__(256)
void k_ln_inplace(float* __restrict__ X, const float* __restrict__ g,
                  const float* __restrict__ bta, int C) {
  float* x = X + (size_t)blockIdx.x * C;
  __shared__ float red[256];
  int tid = threadIdx.x;
  float s = 0.f;
  for (int i = tid; i < C; i += 256) s += x[i];
  red[tid] = s; __syncthreads();
  for (int st = 128; st > 0; st >>= 1) { if (tid < st) red[tid] += red[tid+st]; __syncthreads(); }
  float mean = red[0] / C; __syncthreads();
  float s2 = 0.f;
  for (int i = tid; i < C; i += 256) { float d = x[i] - mean; s2 += d*d; }
  red[tid] = s2; __syncthreads();
  for (int st = 128; st > 0; st >>= 1) { if (tid < st) red[tid] += red[tid+st]; __syncthreads(); }
  float rstd = rsqrtf(red[0] / C + 1e-5f);
  for (int i = tid; i < C; i += 256) x[i] = (x[i] - mean) * rstd * g[i] + bta[i];
}

// 7b) LayerNorm per row, f16 output
__global__ __launch_bounds__(256)
void k_ln_h16(const float* __restrict__ X, const float* __restrict__ g,
              const float* __restrict__ bta, _Float16* __restrict__ Y, int C) {
  const float* x = X + (size_t)blockIdx.x * C;
  _Float16* y = Y + (size_t)blockIdx.x * C;
  __shared__ float red[256];
  int tid = threadIdx.x;
  float s = 0.f;
  for (int i = tid; i < C; i += 256) s += x[i];
  red[tid] = s; __syncthreads();
  for (int st = 128; st > 0; st >>= 1) { if (tid < st) red[tid] += red[tid+st]; __syncthreads(); }
  float mean = red[0] / C; __syncthreads();
  float s2 = 0.f;
  for (int i = tid; i < C; i += 256) { float d = x[i] - mean; s2 += d*d; }
  red[tid] = s2; __syncthreads();
  for (int st = 128; st > 0; st >>= 1) { if (tid < st) red[tid] += red[tid+st]; __syncthreads(); }
  float rstd = rsqrtf(red[0] / C + 1e-5f);
  for (int i = tid; i < C; i += 256) y[i] = (_Float16)((x[i] - mean) * rstd * g[i] + bta[i]);
}

// ---------------------------------------------------------------------------
// 8) Mean over 256 tokens -> f16 [128,768]
// ---------------------------------------------------------------------------
__global__ __launch_bounds__(256)
void k_mean(const float* __restrict__ tok1, _Float16* __restrict__ meanh) {
  int i = blockIdx.x * 256 + threadIdx.x;   // < 98304
  int n = i / 768, f = i % 768;
  float s = 0.f;
  for (int t = 0; t < 256; ++t) s += tok1[((size_t)n * 256 + t) * 768 + f];
  meanh[i] = (_Float16)(s * (1.f / 256.f));
}

// 9) feat rows (d*B+b) -> cv rows (b*D+d)
__global__ __launch_bounds__(256)
void k_permute(const float* __restrict__ featb, float* __restrict__ cvb) {
  int i = blockIdx.x * 256 + threadIdx.x;   // < 98304
  int f = i % 768, r = i / 768;
  int b = r / DD, d = r % DD;
  cvb[i] = featb[((size_t)(d * BB + b)) * 768 + f];
}

// ---------------------------------------------------------------------------
// 10) Attention: per (b,head), seq D=64, head dim 96.
//     Q/V head panels staged to LDS via async global->LDS copies when available.
// ---------------------------------------------------------------------------
__global__ __launch_bounds__(64)
void k_attention(const float* __restrict__ q2, const float* __restrict__ k2,
                 const float* __restrict__ v2, float* __restrict__ o2) {
  __shared__ float qs[DD * HDN];
  __shared__ float vs[DD * HDN];
  int b = blockIdx.x / NHN, h = blockIdx.x % NHN;
  int tid = threadIdx.x;
  for (int i = tid * 4; i < DD * HDN; i += 256) {
    int row = i / HDN, e = i % HDN;
    size_t goff = (size_t)(b * DD + row) * HIDN + h * HDN + e;
#if HAS_ASYNC_LDS
    __builtin_amdgcn_global_load_async_to_lds_b128(
        (as1_i32x4*)(i32x4*)(q2 + goff),
        (as3_i32x4*)(i32x4*)(&qs[i]), 0, 0);
    __builtin_amdgcn_global_load_async_to_lds_b128(
        (as1_i32x4*)(i32x4*)(v2 + goff),
        (as3_i32x4*)(i32x4*)(&vs[i]), 0, 0);
#else
    #pragma unroll
    for (int j = 0; j < 4; ++j) { qs[i + j] = q2[goff + j]; vs[i + j] = v2[goff + j]; }
#endif
  }
#if HAS_ASYNC_LDS
  __builtin_amdgcn_s_wait_asynccnt(0);
#endif
  __syncthreads();
  int qd = tid;
  float s[DD];
  float mx = -1e30f;
  const float rs = 0.1020620726f;   // 1/sqrt(96)
  for (int kd = 0; kd < DD; ++kd) {
    const float* kr = k2 + ((size_t)(b * DD + kd)) * HIDN + h * HDN;
    float acc = 0.f;
    for (int e = 0; e < HDN; ++e) acc += qs[qd * HDN + e] * kr[e];
    s[kd] = acc * rs;
    mx = fmaxf(mx, s[kd]);
  }
  float ssum = 0.f;
  #pragma unroll
  for (int kd = 0; kd < DD; ++kd) { s[kd] = expf(s[kd] - mx); ssum += s[kd]; }
  float inv = 1.f / ssum;
  for (int e = 0; e < HDN; ++e) {
    float acc = 0.f;
    #pragma unroll
    for (int kd = 0; kd < DD; ++kd) acc += s[kd] * vs[kd * HDN + e];
    o2[((size_t)(b * DD + qd)) * HIDN + h * HDN + e] = acc * inv;
  }
}

// ---------------------------------------------------------------------------
// 11) Softmax over depth, expected depth, 4-window confidence
// ---------------------------------------------------------------------------
__global__ __launch_bounds__(256)
void k_softmax_depth(float* __restrict__ prob, const float* __restrict__ dvv,
                     float* __restrict__ depth, float* __restrict__ conf) {
  int pix = blockIdx.x * 256 + threadIdx.x;
  int b = blockIdx.y;
  float p[DD];
  float mx = -1e30f;
  #pragma unroll
  for (int d = 0; d < DD; ++d) {
    p[d] = prob[((size_t)(b * DD + d)) * HWp + pix];
    mx = fmaxf(mx, p[d]);
  }
  float s = 0.f;
  #pragma unroll
  for (int d = 0; d < DD; ++d) { p[d] = expf(p[d] - mx); s += p[d]; }
  float inv = 1.f / s;
  float de = 0.f, di = 0.f;
  #pragma unroll
  for (int d = 0; d < DD; ++d) {
    p[d] *= inv;
    prob[((size_t)(b * DD + d)) * HWp + pix] = p[d];
    de += p[d] * dvv[b * DD + d];
    di += p[d] * (float)d;
  }
  int idx = (int)di;
  float c = 0.f;
  #pragma unroll
  for (int d = 0; d < DD; ++d)
    if (d >= idx - 1 && d <= idx + 2) c += p[d];
  depth[(size_t)b * HWp + pix] = de;
  conf[(size_t)b * HWp + pix]  = c;
}

// ---------------------------------------------------------------------------
// 12) U-Net helpers
// ---------------------------------------------------------------------------
__global__ __launch_bounds__(256)
void k_extract_ref(const float* __restrict__ imgs, float* __restrict__ x0) {
  int i = blockIdx.x * 256 + threadIdx.x;   // < 2*3*HWp
  int b = i / (3 * HWp), r = i % (3 * HWp);
  x0[i] = imgs[(size_t)b * VV * 3 * HWp + r];
}

__global__ __launch_bounds__(256)
void k_conv3x3(const float* __restrict__ in, const float* __restrict__ w,
               const float* __restrict__ bias, float* __restrict__ out,
               int N, int Cin, int Cout, int Hin, int Win, int stride, int relu) {
  int Ho = Hin / stride, Wo = Win / stride;
  long total = (long)N * Cout * Ho * Wo;
  long idx = (long)blockIdx.x * 256 + threadIdx.x;
  if (idx >= total) return;
  int ox = (int)(idx % Wo);
  int oy = (int)((idx / Wo) % Ho);
  int co = (int)((idx / ((long)Wo * Ho)) % Cout);
  int n  = (int)(idx / ((long)Wo * Ho * Cout));
  int pl = (stride == 1) ? 1 : 0;
  float acc = bias[co];
  for (int ci = 0; ci < Cin; ++ci) {
    const float* base  = in + ((size_t)(n * Cin + ci)) * Hin * Win;
    const float* wbase = w + ((size_t)(co * Cin + ci)) * 9;
    for (int ky = 0; ky < 3; ++ky) {
      int iy = oy * stride + ky - pl;
      if (iy < 0 || iy >= Hin) continue;
      for (int kx = 0; kx < 3; ++kx) {
        int ix = ox * stride + kx - pl;
        if (ix < 0 || ix >= Win) continue;
        acc += base[(size_t)iy * Win + ix] * wbase[ky * 3 + kx];
      }
    }
  }
  if (relu) acc = fmaxf(acc, 0.f);
  out[idx] = acc;
}

__global__ __launch_bounds__(256)
void k_concat_up(const float* __restrict__ a, const float* __restrict__ bsrc,
                 float* __restrict__ out, int N, int Ca, int Cb, int Ho, int Wo) {
  long total = (long)N * (Ca + Cb) * Ho * Wo;
  long i = (long)blockIdx.x * 256 + threadIdx.x;
  if (i >= total) return;
  int x = (int)(i % Wo);
  int y = (int)((i / Wo) % Ho);
  int c = (int)((i / ((long)Wo * Ho)) % (Ca + Cb));
  int n = (int)(i / ((long)Wo * Ho * (Ca + Cb)));
  float v;
  if (c < Ca)
    v = a[((size_t)(n * Ca + c)) * (Ho/2) * (Wo/2) + (size_t)(y/2) * (Wo/2) + (x/2)];
  else
    v = bsrc[((size_t)(n * Cb + (c - Ca))) * Ho * Wo + (size_t)y * Wo + x];
  out[i] = v;
}

__global__ __launch_bounds__(256)
void k_conv1x1_sig(const float* __restrict__ in, const float* __restrict__ w,
                   const float* __restrict__ b0, float* __restrict__ out) {
  int i = blockIdx.x * 256 + threadIdx.x;   // < 2*HWp
  int n = i / HWp, p = i % HWp;
  float s = b0[0];
  #pragma unroll
  for (int c = 0; c < 16; ++c) s += w[c] * in[((size_t)(n * 16 + c)) * HWp + p];
  out[i] = 1.f / (1.f + expf(-s));
}

// ---------------------------------------------------------------------------
// Host launcher
// ---------------------------------------------------------------------------
extern "C" void kernel_launch(void* const* d_in, const int* in_sizes, int n_in,
                              void* d_out, int out_size, void* d_ws, size_t ws_size,
                              hipStream_t stream) {
  (void)in_sizes; (void)n_in; (void)out_size; (void)ws_size;
  const float* imgs   = (const float*)d_in[0];
  const float* pm     = (const float*)d_in[1];
  const float* dvv    = (const float*)d_in[2];
  const float* gate_w1= (const float*)d_in[3];
  const float* gate_b1= (const float*)d_in[4];
  const float* gate_w2= (const float*)d_in[5];
  const float* gate_b2= (const float*)d_in[6];
  const float* pe_w   = (const float*)d_in[7];
  const float* pe_b   = (const float*)d_in[8];
  const float* fln_g  = (const float*)d_in[9];
  const float* fln_b  = (const float*)d_in[10];
  const float* feat_w = (const float*)d_in[11];
  const float* feat_b = (const float*)d_in[12];
  const float* ln1_g  = (const float*)d_in[13];
  const float* ln1_b  = (const float*)d_in[14];
  const float* wq     = (const float*)d_in[15];
  const float* wk     = (const float*)d_in[16];
  const float* wv     = (const float*)d_in[17];
  const float* wo     = (const float*)d_in[18];
  const float* ln2_g  = (const float*)d_in[19];
  const float* ln2_b  = (const float*)d_in[20];
  const float* ff_w1  = (const float*)d_in[21];
  const float* ff_b1  = (const float*)d_in[22];
  const float* ff_w2  = (const float*)d_in[23];
  const float* ff_b2  = (const float*)d_in[24];
  const float* dec_w  = (const float*)d_in[25];
  const float* dec_b  = (const float*)d_in[26];
  const float* ue1w = (const float*)d_in[27]; const float* ue1b = (const float*)d_in[28];
  const float* ue2w = (const float*)d_in[29]; const float* ue2b = (const float*)d_in[30];
  const float* ue3w = (const float*)d_in[31]; const float* ue3b = (const float*)d_in[32];
  const float* ud2w = (const float*)d_in[33]; const float* ud2b = (const float*)d_in[34];
  const float* ud1w = (const float*)d_in[35]; const float* ud1b = (const float*)d_in[36];
  const float* uow  = (const float*)d_in[37]; const float* uob  = (const float*)d_in[38];

  float* outF = (float*)d_out;
  float* out_depth = outF;
  float* out_prob  = outF + (size_t)BB * HWp;
  float* out_sem   = out_prob + (size_t)BB * DD * HWp;
  float* out_conf  = out_sem + (size_t)BB * HWp;

  float* ws = (float*)d_ws;
  size_t off = 0;
  auto alloc = [&](size_t n) -> float* {
    float* p = ws + off; off += (n + 63) & ~(size_t)63; return p;
  };
  float* prt  = alloc(128);
  float* xbuf = alloc(25165824);          // diff volume; reused as tok1
  float* accv = alloc(25165824);          // cost volume; reused as U-Net pool
  _Float16* tokh    = (_Float16*)alloc(12582912);
  _Float16* pe_wh   = (_Float16*)alloc(294912);
  _Float16* feat_wh = (_Float16*)alloc(294912);
  _Float16* wqh  = (_Float16*)alloc(294912);
  _Float16* wkh  = (_Float16*)alloc(294912);
  _Float16* wvh  = (_Float16*)alloc(294912);
  _Float16* woh  = (_Float16*)alloc(294912);
  _Float16* ff1h = (_Float16*)alloc(1179648);
  _Float16* ff2h = (_Float16*)alloc(1179648);
  _Float16* decwh = (_Float16*)alloc(25165824);
  _Float16* meanh = (_Float16*)alloc(49152);
  float* featb = alloc(98304);
  float* cvb   = alloc(98304);
  _Float16* h16 = (_Float16*)alloc(49152);
  float* q2  = alloc(98304);
  float* k2  = alloc(98304);
  float* v2b = alloc(98304);
  float* o2  = alloc(98304);
  _Float16* o2h = (_Float16*)alloc(49152);
  float* ffb = alloc(393216);
  _Float16* ffh = (_Float16*)alloc(196608);
  _Float16* cvh = (_Float16*)alloc(49152);
  float* tok1 = xbuf;                     // alias: xbuf free after gate stage
  // U-Net pool aliases accv (free after patchify)
  float* u = accv;
  float* x0   = u; u += 393216;
  float* e1   = u; u += 2097152;
  float* e2   = u; u += 1048576;
  float* e3   = u; u += 524288;
  float* cat2 = u; u += 3145728;
  float* d2b  = u; u += 1048576;
  float* cat1 = u; u += 6291456;
  float* d1b  = u; u += 2097152;

  auto cvt = [&](const float* s, _Float16* d, long n) {
    k_f2h<<<dim3((unsigned)((n + 255) / 256)), 256, 0, stream>>>(s, d, n);
  };
  auto packB = [&](const float* s, _Float16* d, int K, int N) {
    long n = (long)K * N;
    k_packB<<<dim3((unsigned)((n + 255) / 256)), 256, 0, stream>>>(s, d, K, N);
  };
  auto gemm = [&](const _Float16* A, const _Float16* Bp, const float* bias,
                  const float* Cin, float* Cout, int M, int N, int K, int flags) {
    k_gemm<<<dim3(N / 128, M / 16), 128, 0, stream>>>(A, Bp, bias, Cin, Cout, M, N, K, flags);
  };
  auto conv = [&](const float* in, const float* w, const float* bb, float* out,
                  int N, int Cin, int Cout, int Hin, int Win, int stride, int relu) {
    long total = (long)N * Cout * (Hin / stride) * (Win / stride);
    k_conv3x3<<<dim3((unsigned)((total + 255) / 256)), 256, 0, stream>>>(
        in, w, bb, out, N, Cin, Cout, Hin, Win, stride, relu);
  };

  // --- Cost volume ---
  k_proj<<<1, 32, 0, stream>>>(pm, prt);
  for (int v = 1; v < VV; ++v) {
    k_warpdiff<<<dim3(256, 128), 256, 0, stream>>>(imgs, prt, dvv, xbuf, v);
    k_gate<<<dim3(256, 128), 256, 0, stream>>>(xbuf, gate_w1, gate_b1, gate_w2, gate_b2,
                                               accv, (v == 1) ? 1 : 0);
  }
  k_patchify<<<98304, 256, 0, stream>>>(accv, tokh);

  // --- Weight conversions to packed-f16 panels ---
  packB(pe_w,   pe_wh,   768, 768);
  packB(feat_w, feat_wh, 768, 768);
  packB(wq, wqh, 768, 768);
  packB(wk, wkh, 768, 768);
  packB(wv, wvh, 768, 768);
  packB(wo, woh, 768, 768);
  packB(ff_w1, ff1h, 768, 3072);
  packB(ff_w2, ff2h, 3072, 768);
  packB(dec_w, decwh, 768, 65536);

  // --- Patch embed + feature head ---
  gemm(tokh, pe_wh, pe_b, nullptr, tok1, 32768, 768, 768, GF_BIAS);
  k_ln_inplace<<<32768, 256, 0, stream>>>(tok1, fln_g, fln_b, 768);
  k_mean<<<384, 256, 0, stream>>>(tok1, meanh);
  gemm(meanh, feat_wh, feat_b, nullptr, featb, 128, 768, 768, GF_BIAS);
  k_permute<<<384, 256, 0, stream>>>(featb, cvb);

  // --- Transformer block over depth dim ---
  k_ln_h16<<<128, 256, 0, stream>>>(cvb, ln1_g, ln1_b, h16, 768);
  gemm(h16, wqh, nullptr, nullptr, q2, 128, 768, 768, 0);
  gemm(h16, wkh, nullptr, nullptr, k2, 128, 768, 768, 0);
  gemm(h16, wvh, nullptr, nullptr, v2b, 128, 768, 768, 0);
  k_attention<<<16, 64, 0, stream>>>(q2, k2, v2b, o2);
  cvt(o2, o2h, 98304);
  gemm(o2h, woh, nullptr, cvb, cvb, 128, 768, 768, GF_BETA);
  k_ln_h16<<<128, 256, 0, stream>>>(cvb, ln2_g, ln2_b, h16, 768);
  gemm(h16, ff1h, ff_b1, nullptr, ffb, 128, 3072, 768, GF_BIAS | GF_RELU);
  cvt(ffb, ffh, 393216);
  gemm(ffh, ff2h, ff_b2, cvb, cvb, 128, 768, 3072, GF_BIAS | GF_BETA);

  // --- Decoder + softmax/depth/conf ---
  cvt(cvb, cvh, 98304);
  gemm(cvh, decwh, dec_b, nullptr, out_prob, 128, 65536, 768, GF_BIAS);
  k_softmax_depth<<<dim3(256, 2), 256, 0, stream>>>(out_prob, dvv, out_depth, out_conf);

  // --- U-Net (after patchify, accv safe to reuse) ---
  k_extract_ref<<<1536, 256, 0, stream>>>(imgs, x0);
  conv(x0, ue1w, ue1b, e1, 2, 3, 16, 256, 256, 1, 1);
  conv(e1, ue2w, ue2b, e2, 2, 16, 32, 256, 256, 2, 1);
  conv(e2, ue3w, ue3b, e3, 2, 32, 64, 128, 128, 2, 1);
  k_concat_up<<<12288, 256, 0, stream>>>(e3, e2, cat2, 2, 64, 32, 128, 128);
  conv(cat2, ud2w, ud2b, d2b, 2, 96, 32, 128, 128, 1, 1);
  k_concat_up<<<24576, 256, 0, stream>>>(d2b, e1, cat1, 2, 32, 16, 256, 256);
  conv(cat1, ud1w, ud1b, d1b, 2, 48, 16, 256, 256, 1, 1);
  k_conv1x1_sig<<<512, 256, 0, stream>>>(d1b, uow, uob, out_sem);
}
